// MultiHeadAttentionBlock_81776177316344
// MI455X (gfx1250) — matrix operand order repaired and checked by reading, not compiled, not run on hardware
//
#include <hip/hip_runtime.h>

// MI455X (gfx1250, wave32) multi-head attention block.
// GEMMs + attention on v_wmma_f32_16x16x32_f16 (f16 in, f32 acc), with
// double-buffered async global->LDS staging (ASYNCcnt) for shared tiles.

typedef _Float16 v16h __attribute__((ext_vector_type(16)));
typedef _Float16 v8h  __attribute__((ext_vector_type(8)));
typedef _Float16 v4h  __attribute__((ext_vector_type(4)));
typedef float    v8f  __attribute__((ext_vector_type(8)));
typedef int      v4i  __attribute__((ext_vector_type(4)));

#define H   1024
#define SEQ 2048
#define BB  2
#define NH  16
#define HD  64
#define MTOT (BB * SEQ)   // 4096 token rows

#if defined(__AMDGCN__) && __has_builtin(__builtin_amdgcn_global_load_async_to_lds_b128)
#define USE_ASYNC_LDS 1
#else
#define USE_ASYNC_LDS 0
#endif

__device__ __forceinline__ v8f wmma_f16(v16h a, v16h b, v8f c) {
    // D = A(16x32 f16) x B(32x16 f16) + C(16x16 f32)
    return __builtin_amdgcn_wmma_f32_16x16x32_f16(
        false, a, false, b, (short)0, c, false, false);
}

// A-fragment: 16 contiguous K halves (per half-wave, K = 16*half + s).
__device__ __forceinline__ v16h make_afrag(const _Float16* p) {
    v8h lo = *(const v8h*)p;
    v8h hi = *(const v8h*)(p + 8);
    return __builtin_shufflevector(lo, hi, 0,1,2,3,4,5,6,7,8,9,10,11,12,13,14,15);
}

// 16-byte global -> LDS copy: async DMA (ASYNCcnt) when available.
__device__ __forceinline__ void copy16_g2l(const _Float16* g, _Float16* l) {
#if USE_ASYNC_LDS
    __builtin_amdgcn_global_load_async_to_lds_b128(
        (__attribute__((address_space(1))) v4i*)(g),
        (__attribute__((address_space(3))) v4i*)(l), 0, 0);
#else
    *(v8h*)l = *(const v8h*)g;
#endif
}

__device__ __forceinline__ void wait_async_lds() {
#if USE_ASYNC_LDS
#if __has_builtin(__builtin_amdgcn_s_wait_asynccnt)
    __builtin_amdgcn_s_wait_asynccnt(0);
#else
    asm volatile("s_wait_asynccnt 0x0" ::: "memory");
#endif
#endif
}

// ---------------------------------------------------------------- f32 -> f16
__global__ void cvt_f32_to_f16_kernel(const float* __restrict__ in,
                                      _Float16* __restrict__ out, int n4) {
    int i = blockIdx.x * blockDim.x + threadIdx.x;
    if (i < n4) {
        float4 v = ((const float4*)in)[i];
        v4h o; o[0] = (_Float16)v.x; o[1] = (_Float16)v.y;
               o[2] = (_Float16)v.z; o[3] = (_Float16)v.w;
        ((v4h*)out)[i] = o;
    }
}

// ------------------------------------------------------- fused LayerNorm->f16
__global__ __launch_bounds__(256) void ln_kernel(
    const float* __restrict__ x, const float* __restrict__ g,
    const float* __restrict__ be, _Float16* __restrict__ xn) {
    const int row = blockIdx.x;
    const int tid = threadIdx.x;
    const float4 v = ((const float4*)(x + (size_t)row * H))[tid];

    float s = v.x + v.y + v.z + v.w;
    #pragma unroll
    for (int m = 16; m >= 1; m >>= 1) s += __shfl_xor(s, m, 32);
    __shared__ float r1[8], r2[8];
    if ((tid & 31) == 0) r1[tid >> 5] = s;
    __syncthreads();
    float mu = 0.f;
    #pragma unroll
    for (int i = 0; i < 8; ++i) mu += r1[i];
    mu *= (1.0f / (float)H);

    const float d0 = v.x - mu, d1 = v.y - mu, d2 = v.z - mu, d3 = v.w - mu;
    float s2 = d0*d0 + d1*d1 + d2*d2 + d3*d3;
    #pragma unroll
    for (int m = 16; m >= 1; m >>= 1) s2 += __shfl_xor(s2, m, 32);
    if ((tid & 31) == 0) r2[tid >> 5] = s2;
    __syncthreads();
    float var = 0.f;
    #pragma unroll
    for (int i = 0; i < 8; ++i) var += r2[i];
    var *= (1.0f / (float)H);
    const float rstd = rsqrtf(var + 1e-5f);

    const int c = tid * 4;
    v4h o;
    o[0] = (_Float16)(d0 * rstd * g[c + 0] + be[c + 0]);
    o[1] = (_Float16)(d1 * rstd * g[c + 1] + be[c + 1]);
    o[2] = (_Float16)(d2 * rstd * g[c + 2] + be[c + 2]);
    o[3] = (_Float16)(d3 * rstd * g[c + 3] + be[c + 3]);
    *(v4h*)(xn + (size_t)row * H + c) = o;
}

// Stage a 128x32-half A tile into LDS (rows padded to 40 halves).
// 512 16B chunks / 256 threads = 2 per thread.
__device__ __forceinline__ void stage_a_tile(const _Float16* __restrict__ A,
                                             int rowA0, int k0,
                                             _Float16* lbuf, int tid) {
    #pragma unroll
    for (int i = 0; i < 2; ++i) {
        const int ch  = tid * 2 + i;
        const int row = ch >> 2;
        const int cc  = ch & 3;
        copy16_g2l(A + (size_t)(rowA0 + row) * H + k0 + cc * 8,
                   lbuf + row * 40 + cc * 8);
    }
}

// ------------------------------------------------------------------ QKV GEMM
// qkv[m,n] = xn[m,:] . Wqkv[n,:] + b[n].  Block tile 128x256, wave tile 64x64.
// A tile async-staged in LDS (shared by the 4 waves of each wave-row).
// Epilogue scatters Q (scaled 1/8) / K head-major, V transposed [bh][d][s].
__global__ __launch_bounds__(256) void qkv_gemm_kernel(
    const _Float16* __restrict__ A, const _Float16* __restrict__ Bw,
    const float* __restrict__ bias,
    _Float16* __restrict__ qbuf, _Float16* __restrict__ kbuf,
    _Float16* __restrict__ vtbuf) {
    const int tid   = threadIdx.x;
    const int lane  = tid & 31;
    const int w     = tid >> 5;
    const int lm    = lane & 15;
    const int khalf = lane >> 4;
    const int rowA0 = blockIdx.y * 128;
    const int lrow0 = (w >> 2) * 64;
    const int row0  = rowA0 + lrow0;
    const int col0  = blockIdx.x * 256 + (w & 3) * 64;

    __shared__ __align__(16) _Float16 Asm[2][128 * 40];

    stage_a_tile(A, rowA0, 0, Asm[0], tid);

    v8f acc[4][4] = {};
    for (int step = 0; step < H / 32; ++step) {
        const int cur = step & 1;
        const int k0  = step * 32;
        wait_async_lds();
        __syncthreads();
        if (step + 1 < H / 32)
            stage_a_tile(A, rowA0, k0 + 32, Asm[cur ^ 1], tid);

        v16h af[4], bf[4];
        #pragma unroll
        for (int mi = 0; mi < 4; ++mi)
            af[mi] = make_afrag(&Asm[cur][(lrow0 + mi * 16 + lm) * 40 + khalf * 16]);
        #pragma unroll
        for (int ni = 0; ni < 4; ++ni)
            bf[ni] = *(const v16h*)(Bw + (size_t)(col0 + ni * 16 + lm) * H + k0 + khalf * 16);
        #pragma unroll
        for (int mi = 0; mi < 4; ++mi)
            #pragma unroll
            for (int ni = 0; ni < 4; ++ni)
                acc[mi][ni] = wmma_f16(af[mi], bf[ni], acc[mi][ni]);
    }

    #pragma unroll
    for (int ni = 0; ni < 4; ++ni) {
        const int n     = col0 + ni * 16 + lm;
        const int which = n >> 10;          // 0=Q 1=K 2=V (uniform per tile)
        const int rem   = n & 1023;
        const int head  = rem >> 6;
        const int d     = rem & 63;
        const float bv  = bias[n];
        #pragma unroll
        for (int mi = 0; mi < 4; ++mi) {
            const int m0  = row0 + mi * 16 + khalf * 8;   // 8 consecutive rows
            const int bid = m0 >> 11;
            const int s0  = m0 & 2047;
            const int bh  = bid * NH + head;
            if (which == 2) {
                v8h pk;
                #pragma unroll
                for (int r = 0; r < 8; ++r) pk[r] = (_Float16)(acc[mi][ni][r] + bv);
                *(v8h*)(vtbuf + ((size_t)bh * HD + d) * SEQ + s0) = pk;
            } else {
                _Float16* dst = (which == 0 ? qbuf : kbuf) +
                                ((size_t)bh * SEQ + s0) * HD + d;
                const float sc = (which == 0) ? 0.125f : 1.0f;  // 1/sqrt(HD) into Q
                #pragma unroll
                for (int r = 0; r < 8; ++r)
                    dst[(size_t)r * HD] = (_Float16)((acc[mi][ni][r] + bv) * sc);
            }
        }
    }
}

// ---------------------------------------------------- flash attention (wave32)
// Block = 8 waves, all sharing one (b,head); each wave owns a 16-query tile.
// K/V 32-key chunks are async-staged into LDS once per block (8x reuse),
// double buffered.  Scores: 4 WMMA; online softmax f32; P transposed through
// per-wave LDS; ctx: 4 WMMA.
__global__ __launch_bounds__(256) void attn_kernel(
    const _Float16* __restrict__ qbuf, const _Float16* __restrict__ kbuf,
    const _Float16* __restrict__ vtbuf, _Float16* __restrict__ ctx) {
    const int bh    = blockIdx.x >> 4;
    const int qblk  = blockIdx.x & 15;
    const int tid   = threadIdx.x;
    const int w     = tid >> 5;
    const int lane  = tid & 31;
    const int lm    = lane & 15;
    const int khalf = lane >> 4;
    const int qbase = (qblk * 8 + w) * 16;

    const _Float16* Q = qbuf + ((size_t)bh * SEQ + qbase) * HD;
    const _Float16* K = kbuf + (size_t)bh * SEQ * HD;
    const _Float16* V = vtbuf + (size_t)bh * HD * SEQ;

    __shared__ __align__(16) _Float16 Klds[2][32 * 72];   // 32 keys x 64 d (+8 pad)
    __shared__ __align__(16) _Float16 Vlds[2][64 * 40];   // 64 d x 32 keys (+8 pad)
    __shared__ __align__(16) _Float16 pbuf[8][16][40];

    // stage K/V chunk: K 32x64 halves (256 chunks), V^T 64x32 halves (256 chunks)
    auto stage_kv = [&](int kb, _Float16* kl, _Float16* vl) {
        { const int row = tid >> 3, cc = tid & 7;
          copy16_g2l(K + (size_t)(kb + row) * HD + cc * 8, kl + row * 72 + cc * 8); }
        { const int row = tid >> 2, cc = tid & 3;
          copy16_g2l(V + (size_t)row * SEQ + kb + cc * 8, vl + row * 40 + cc * 8); }
    };

    v16h qf[2];
    #pragma unroll
    for (int f = 0; f < 2; ++f)
        qf[f] = make_afrag(Q + (size_t)lm * HD + f * 32 + khalf * 16);

    v8f acc[4] = {};
    float mrow[8], lrow[8];
    #pragma unroll
    for (int r = 0; r < 8; ++r) { mrow[r] = -1e30f; lrow[r] = 0.f; }

    stage_kv(0, Klds[0], Vlds[0]);

    for (int c = 0; c < SEQ / 32; ++c) {
        const int cur = c & 1;
        const int kb  = c * 32;
        wait_async_lds();
        __syncthreads();                       // K/V[cur] visible; prior reads done
        if (c + 1 < SEQ / 32)
            stage_kv(kb + 32, Klds[cur ^ 1], Vlds[cur ^ 1]);

        // scores: two 16-key sub-tiles, d split into two 32-chunks
        v8f sl = {}, sh = {};
        #pragma unroll
        for (int g = 0; g < 2; ++g) {
            v16h bl = *(const v16h*)&Klds[cur][lm * 72 + g * 32 + khalf * 16];
            sl = wmma_f16(qf[g], bl, sl);
        }
        #pragma unroll
        for (int g = 0; g < 2; ++g) {
            v16h bhf = *(const v16h*)&Klds[cur][(16 + lm) * 72 + g * 32 + khalf * 16];
            sh = wmma_f16(qf[g], bhf, sh);
        }

        // online softmax per row (row lives across a 16-lane half)
        float alpha[8];
        #pragma unroll
        for (int r = 0; r < 8; ++r) {
            float mx = fmaxf(sl[r], sh[r]);
            #pragma unroll
            for (int msk = 8; msk >= 1; msk >>= 1) mx = fmaxf(mx, __shfl_xor(mx, msk, 32));
            const float mnew = fmaxf(mrow[r], mx);
            alpha[r] = __expf(mrow[r] - mnew);
            mrow[r]  = mnew;
            const float pl = __expf(sl[r] - mnew);
            const float ph = __expf(sh[r] - mnew);
            sl[r] = pl; sh[r] = ph;
            float rs = pl + ph;
            #pragma unroll
            for (int msk = 8; msk >= 1; msk >>= 1) rs += __shfl_xor(rs, msk, 32);
            lrow[r] = lrow[r] * alpha[r] + rs;
        }
        #pragma unroll
        for (int t = 0; t < 4; ++t)
            #pragma unroll
            for (int r = 0; r < 8; ++r) acc[t][r] *= alpha[r];

        // transpose P (C-layout -> A-layout) through this wave's LDS region
        #pragma unroll
        for (int r = 0; r < 8; ++r) {
            const int row = r + khalf * 8;
            pbuf[w][row][lm]      = (_Float16)sl[r];
            pbuf[w][row][16 + lm] = (_Float16)sh[r];
        }
        __syncthreads();
        const v16h pf = make_afrag(&pbuf[w][lm][khalf * 16]);

        #pragma unroll
        for (int t = 0; t < 4; ++t) {
            v16h vf = *(const v16h*)&Vlds[cur][(t * 16 + lm) * 40 + khalf * 16];
            acc[t] = wmma_f16(pf, vf, acc[t]);
        }
    }

    float inv[8];
    #pragma unroll
    for (int r = 0; r < 8; ++r) inv[r] = 1.0f / lrow[r];

    const int head = bh & (NH - 1);
    const int bid  = bh >> 4;
    #pragma unroll
    for (int t = 0; t < 4; ++t) {
        const int col = head * HD + t * 16 + lm;
        #pragma unroll
        for (int r = 0; r < 8; ++r) {
            const int row = qbase + r + khalf * 8;
            ctx[((size_t)bid * SEQ + row) * H + col] = (_Float16)(acc[t][r] * inv[r]);
        }
    }
}

// ------------------------------------------- output projection + bias + residual
__global__ __launch_bounds__(256) void out_gemm_kernel(
    const _Float16* __restrict__ A, const _Float16* __restrict__ Bw,
    const float* __restrict__ bias, const float* __restrict__ resid,
    float* __restrict__ out) {
    const int tid   = threadIdx.x;
    const int lane  = tid & 31;
    const int w     = tid >> 5;
    const int lm    = lane & 15;
    const int khalf = lane >> 4;
    const int rowA0 = blockIdx.y * 128;
    const int lrow0 = (w >> 2) * 64;
    const int row0  = rowA0 + lrow0;
    const int col0  = blockIdx.x * 256 + (w & 3) * 64;

    __shared__ __align__(16) _Float16 Asm[2][128 * 40];

    stage_a_tile(A, rowA0, 0, Asm[0], tid);

    v8f acc[4][4] = {};
    for (int step = 0; step < H / 32; ++step) {
        const int cur = step & 1;
        const int k0  = step * 32;
        wait_async_lds();
        __syncthreads();
        if (step + 1 < H / 32)
            stage_a_tile(A, rowA0, k0 + 32, Asm[cur ^ 1], tid);

        v16h af[4], bf[4];
        #pragma unroll
        for (int mi = 0; mi < 4; ++mi)
            af[mi] = make_afrag(&Asm[cur][(lrow0 + mi * 16 + lm) * 40 + khalf * 16]);
        #pragma unroll
        for (int ni = 0; ni < 4; ++ni)
            bf[ni] = *(const v16h*)(Bw + (size_t)(col0 + ni * 16 + lm) * H + k0 + khalf * 16);
        #pragma unroll
        for (int mi = 0; mi < 4; ++mi)
            #pragma unroll
            for (int ni = 0; ni < 4; ++ni)
                acc[mi][ni] = wmma_f16(af[mi], bf[ni], acc[mi][ni]);
    }

    #pragma unroll
    for (int ni = 0; ni < 4; ++ni) {
        const int n    = col0 + ni * 16 + lm;
        const float bv = bias[n];
        #pragma unroll
        for (int mi = 0; mi < 4; ++mi) {
            const int m0 = row0 + mi * 16 + khalf * 8;
            #pragma unroll
            for (int r = 0; r < 8; ++r) {
                const size_t idx = (size_t)(m0 + r) * H + n;
                out[idx] = acc[mi][ni][r] + bv + resid[idx];
            }
        }
    }
}

// ---------------------------------------------------------------------- launch
extern "C" void kernel_launch(void* const* d_in, const int* in_sizes, int n_in,
                              void* d_out, int out_size, void* d_ws, size_t ws_size,
                              hipStream_t stream) {
    const float* x     = (const float*)d_in[0];
    const float* gamma = (const float*)d_in[1];
    const float* beta  = (const float*)d_in[2];
    const float* Wqkv  = (const float*)d_in[3];
    const float* bqkv  = (const float*)d_in[4];
    const float* Wout  = (const float*)d_in[5];
    const float* bout  = (const float*)d_in[6];
    float* out = (float*)d_out;

    // Workspace layout (f16), total 48 MiB.
    char* ws = (char*)d_ws;
    _Float16* xn     = (_Float16*)(ws);                                // 8 MiB
    _Float16* wqkv_h = (_Float16*)(ws + (8u << 20));                   // 6 MiB
    _Float16* wout_h = (_Float16*)(ws + (14u << 20));                  // 2 MiB
    _Float16* qbuf   = (_Float16*)(ws + (16u << 20));                  // 8 MiB
    _Float16* kbuf   = (_Float16*)(ws + (24u << 20));                  // 8 MiB
    _Float16* vtbuf  = (_Float16*)(ws + (32u << 20));                  // 8 MiB
    _Float16* ctx    = (_Float16*)(ws + (40u << 20));                  // 8 MiB

    // 1) weights -> f16
    cvt_f32_to_f16_kernel<<<(3 * H * H / 4 + 255) / 256, 256, 0, stream>>>(Wqkv, wqkv_h, 3 * H * H / 4);
    cvt_f32_to_f16_kernel<<<(H * H / 4 + 255) / 256, 256, 0, stream>>>(Wout, wout_h, H * H / 4);
    // 2) LayerNorm -> f16 activations
    ln_kernel<<<MTOT, 256, 0, stream>>>(x, gamma, beta, xn);
    // 3) QKV projection (WMMA GEMM, async LDS A-tiles) + head-major scatter
    qkv_gemm_kernel<<<dim3(3 * H / 256, MTOT / 128), 256, 0, stream>>>(
        xn, wqkv_h, bqkv, qbuf, kbuf, vtbuf);
    // 4) flash attention (async LDS K/V staging, WMMA scores + ctx)
    attn_kernel<<<BB * NH * (SEQ / 128), 256, 0, stream>>>(qbuf, kbuf, vtbuf, ctx);
    // 5) output projection + bias + residual
    out_gemm_kernel<<<dim3(H / 256, MTOT / 128), 256, 0, stream>>>(
        ctx, wout_h, bout, x, out);
}
